// GINENet_72035191488560
// MI455X (gfx1250) — matrix operand order repaired and checked by reading, not compiled, not run on hardware
//
#include <hip/hip_runtime.h>
#include <hip/hip_bf16.h>
#include <stdint.h>

// Problem sizes (match reference)
#define N_NODES 100000
#define N_EDGES 1600000
#define G_GRAPHS 2048
#define DIM 32
#define FXD 78
#define FE 7
#define OD 128
#define SEQ 1000
#define VOCAB 26
#define NF 32
#define KW 8
#define CONV_LEN 121
#define SROW (VOCAB * NF * KW) /* 6656 */
#define BN_EPS 1e-5f

typedef __attribute__((ext_vector_type(16))) __bf16 v16bf;
typedef __attribute__((ext_vector_type(8))) float v8f;

struct U16x16 { uint4 lo, hi; };  // 32B == v16bf

__device__ __forceinline__ unsigned short f2bf(float f) {
  unsigned int u = __builtin_bit_cast(unsigned int, f);
  u += 0x7fffu + ((u >> 16) & 1u);  // round-to-nearest-even
  return (unsigned short)(u >> 16);
}

// ---------------------------------------------------------------- utilities
__global__ void fill0(float* __restrict__ p, int n) {
  int i = blockIdx.x * 256 + threadIdx.x;
  if (i < n) p[i] = 0.f;
}

// Pre-swizzle row-major f32 B [K,N] into per-lane WMMA fragment order (bf16).
// B fragment element e of lane L for (n-tile nt, k-tile kt):
//   col = nt*16 + (L&15),  k = kt*32 + (L>>4)*8 + e + (e>=8 ? 8 : 0)   (ISA 7.12.2 B layout)
// Output index: ((nt*nkt + kt)*32 + L)*16 + e  -> one 32B load per lane per k-step.
__global__ void swizzle_B(const float* __restrict__ B, unsigned short* __restrict__ out,
                          int K, int N) {
  int idx = blockIdx.x * 256 + threadIdx.x;
  if (idx >= K * N) return;
  int e = idx & 15;
  int L = (idx >> 4) & 31;
  int tt = idx >> 9;
  int nkt = K >> 5;
  int kt = tt % nkt;
  int nt = tt / nkt;
  int col = (nt << 4) + (L & 15);
  int k = (kt << 5) + ((e < 8) ? (((L >> 4) << 3) + e) : (16 + ((L >> 4) << 3) + (e - 8)));
  out[idx] = f2bf(B[(size_t)k * N + col]);
}

// Pre-swizzle row-major f32 A [M,K] (row stride ld) into A-fragment order (bf16).
// A fragment element e of lane L for (m-tile mt, k-tile kt):
//   row = mt*16 + (L&15), kb = (L>>4)*8,
//   e in [0,8):  k = kt*32 + kb + e
//   e in [8,16): k = kt*32 + 16 + kb + (e-8)               (ISA 7.12.2 A layout)
// Output index: ((mt*nkt + kt)*32 + L)*16 + e
__global__ void swizzle_A(const float* __restrict__ A, unsigned short* __restrict__ out,
                          int K, int ld, int total /* M*K */) {
  int idx = blockIdx.x * 256 + threadIdx.x;
  if (idx >= total) return;
  int e = idx & 15;
  int L = (idx >> 4) & 31;
  int tt = idx >> 9;
  int nkt = K >> 5;
  int kt = tt % nkt;
  int mt = tt / nkt;
  int row = (mt << 4) + (L & 15);
  int kb = (L >> 4) << 3;
  int k = (kt << 5) + ((e < 8) ? (kb + e) : (16 + kb + (e - 8)));
  out[idx] = f2bf(A[(size_t)row * ld + k]);
}

// ------------------------------------------------ WMMA GEMM (bf16 in, f32 acc)
// C[m, coff+n] = act(A@B + bias[n]).  A and B both pre-swizzled to fragment
// order, so the k-loop is pure global_load_b128 + v_wmma (no LDS, no barriers),
// software-pipelined one k-step ahead. Each wave owns a 32x32 output tile
// (2 m-tiles x 2 n-tiles => 4 WMMAs per 4x32B of fragment loads).
// Block = 256 thr = 8 waves laid out 2(m) x 4(n): block tile 64 rows x 128 cols.
// Grid = (N/128, M/64). Requires M%64==0, N%128==0, K%32==0.
template <bool RELU>
__global__ void gemm_bf16_wmma(const unsigned short* __restrict__ Asw,
                               const unsigned short* __restrict__ Bsw,
                               const float* __restrict__ bias, float* __restrict__ C,
                               int K, int N, int ldc, int coff) {
  const int lane = threadIdx.x & 31;
  const int wave = threadIdx.x >> 5;
  const int nkt = K >> 5;
  const int mt0 = blockIdx.y * 4 + (wave >> 2) * 2;  // first of 2 m-tiles
  const int nt0 = blockIdx.x * 8 + (wave & 3) * 2;   // first of 2 n-tiles
  const size_t tileStride = 32 * 16;                 // elems per (tile,k-step)
  const unsigned short* a0 = Asw + (size_t)mt0 * nkt * tileStride + lane * 16;
  const unsigned short* a1 = a0 + (size_t)nkt * tileStride;
  const unsigned short* b0 = Bsw + (size_t)nt0 * nkt * tileStride + lane * 16;
  const unsigned short* b1 = b0 + (size_t)nkt * tileStride;
  v8f acc00 = {0.f, 0.f, 0.f, 0.f, 0.f, 0.f, 0.f, 0.f};
  v8f acc01 = acc00, acc10 = acc00, acc11 = acc00;
  U16x16 pa0, pa1, pb0, pb1;
  pa0.lo = *(const uint4*)(a0); pa0.hi = *(const uint4*)(a0 + 8);
  pa1.lo = *(const uint4*)(a1); pa1.hi = *(const uint4*)(a1 + 8);
  pb0.lo = *(const uint4*)(b0); pb0.hi = *(const uint4*)(b0 + 8);
  pb1.lo = *(const uint4*)(b1); pb1.hi = *(const uint4*)(b1 + 8);
  for (int kt = 0; kt < nkt; ++kt) {
    v16bf A0 = __builtin_bit_cast(v16bf, pa0);
    v16bf A1 = __builtin_bit_cast(v16bf, pa1);
    v16bf B0 = __builtin_bit_cast(v16bf, pb0);
    v16bf B1 = __builtin_bit_cast(v16bf, pb1);
    if (kt + 1 < nkt) {  // prefetch next k-step fragments
      a0 += tileStride; a1 += tileStride; b0 += tileStride; b1 += tileStride;
      pa0.lo = *(const uint4*)(a0); pa0.hi = *(const uint4*)(a0 + 8);
      pa1.lo = *(const uint4*)(a1); pa1.hi = *(const uint4*)(a1 + 8);
      pb0.lo = *(const uint4*)(b0); pb0.hi = *(const uint4*)(b0 + 8);
      pb1.lo = *(const uint4*)(b1); pb1.hi = *(const uint4*)(b1 + 8);
    }
    acc00 = __builtin_amdgcn_wmma_f32_16x16x32_bf16(false, A0, false, B0, (short)0, acc00, false, false);
    acc01 = __builtin_amdgcn_wmma_f32_16x16x32_bf16(false, A0, false, B1, (short)0, acc01, false, false);
    acc10 = __builtin_amdgcn_wmma_f32_16x16x32_bf16(false, A1, false, B0, (short)0, acc10, false, false);
    acc11 = __builtin_amdgcn_wmma_f32_16x16x32_bf16(false, A1, false, B1, (short)0, acc11, false, false);
  }
  // D layout: col = lane&15 within tile, row = r + 8*(lane>>4)
  auto store = [&](v8f acc, int mt, int nt) {
    int col = (nt << 4) + (lane & 15);
    int mb = (mt << 4) + ((lane >> 4) << 3);
    float bv = bias[col];
#pragma unroll
    for (int r = 0; r < 8; ++r) {
      float v = acc[r] + bv;
      if (RELU) v = fmaxf(v, 0.f);
      C[(size_t)(mb + r) * ldc + coff + col] = v;
    }
  };
  store(acc00, mt0, nt0);
  store(acc01, mt0, nt0 + 1);
  store(acc10, mt0 + 1, nt0);
  store(acc11, mt0 + 1, nt0 + 1);
}

// ------------------------------------------------------------- GNN kernels
// Fold layer-0 edge projection: WeL = We @ lin_W  [7,78], beL = be @ lin_W + lin_b
__global__ void prep_wel(const float* __restrict__ We, const float* __restrict__ be,
                         const float* __restrict__ linW, const float* __restrict__ linb,
                         float* __restrict__ WeL, float* __restrict__ beL) {
  int j = threadIdx.x;
  if (j >= FXD) return;
  float bb = linb[j];
  for (int k = 0; k < DIM; ++k) bb += be[k] * linW[k * FXD + j];
  beL[j] = bb;
  for (int f = 0; f < FE; ++f) {
    float s = 0.f;
    for (int k = 0; k < DIM; ++k) s += We[f * DIM + k] * linW[k * FXD + j];
    WeL[f * FXD + j] = s;
  }
}

// One wave per edge, lane per feature dim. ea recomputed on the fly (7 MACs/dim),
// coalesced gather of h[src], relu, f32 atomic scatter-add to agg[dst].
// Node arrays (13-31 MB) are L2-resident, so random access stays on-chip.
__global__ void edge_scatter(const int* __restrict__ ei, const float* __restrict__ eattr,
                             const float* __restrict__ h, const float* __restrict__ W,
                             const float* __restrict__ bvec, float* __restrict__ agg, int d) {
  int e = (blockIdx.x << 3) + (threadIdx.x >> 5);
  int lane = threadIdx.x & 31;
  int src = ei[e];
  int dst = ei[N_EDGES + e];
  const float* ap = eattr + (size_t)e * FE;
  __builtin_prefetch(eattr + (size_t)(e + 64) * FE, 0, 1);  // global_prefetch_b8
  float a0 = ap[0], a1 = ap[1], a2 = ap[2], a3 = ap[3], a4 = ap[4], a5 = ap[5], a6 = ap[6];
  for (int j = lane; j < d; j += 32) {
    float w = bvec[j] + a0 * W[j] + a1 * W[d + j] + a2 * W[2 * d + j] + a3 * W[3 * d + j] +
              a4 * W[4 * d + j] + a5 * W[5 * d + j] + a6 * W[6 * d + j];
    float m = fmaxf(h[(size_t)src * d + j] + w, 0.f);
    atomicAdd(&agg[(size_t)dst * d + j], m);
  }
}

// Per-node 2-layer MLP (weights in LDS), relu, store r=relu(out), accumulate BN
// sums/sumsq via wave-shuffle -> LDS -> global atomics.
template <int IN>
__global__ void node_mlp(const float* __restrict__ hin, const float* __restrict__ agg,
                         const float* __restrict__ W1, const float* __restrict__ b1,
                         const float* __restrict__ W2, const float* __restrict__ b2,
                         float* __restrict__ rout, float* __restrict__ bnstat) {
  __shared__ float sW1[IN * DIM];
  __shared__ float sW2[DIM * DIM];
  __shared__ float sb1[DIM], sb2[DIM];
  __shared__ float red[64];
  int tid = threadIdx.x;
  for (int i = tid; i < IN * DIM; i += 256) sW1[i] = W1[i];
  for (int i = tid; i < DIM * DIM; i += 256) sW2[i] = W2[i];
  if (tid < DIM) { sb1[tid] = b1[tid]; sb2[tid] = b2[tid]; }
  if (tid < 64) red[tid] = 0.f;
  __syncthreads();
  int n = blockIdx.x * 256 + tid;
  bool act = n < N_NODES;
  float accv[DIM];
#pragma unroll
  for (int i = 0; i < DIM; ++i) accv[i] = sb1[i];
  for (int j = 0; j < IN; ++j) {
    float zj = act ? (hin[(size_t)n * IN + j] + agg[(size_t)n * IN + j]) : 0.f;
#pragma unroll
    for (int i = 0; i < DIM; ++i) accv[i] += zj * sW1[j * DIM + i];
  }
  float t[DIM];
#pragma unroll
  for (int i = 0; i < DIM; ++i) t[i] = fmaxf(accv[i], 0.f);
  int lane = tid & 31;
  for (int i = 0; i < DIM; ++i) {
    float m = sb2[i];
#pragma unroll
    for (int k = 0; k < DIM; ++k) m += t[k] * sW2[k * DIM + i];
    float rv = act ? fmaxf(m, 0.f) : 0.f;
    if (act) rout[(size_t)n * DIM + i] = rv;
    float s1 = rv, s2 = rv * rv;
    for (int off = 16; off; off >>= 1) {
      s1 += __shfl_xor(s1, off, 32);
      s2 += __shfl_xor(s2, off, 32);
    }
    if (lane == 0) { atomicAdd(&red[i], s1); atomicAdd(&red[DIM + i], s2); }
  }
  __syncthreads();
  if (tid < 64) atomicAdd(&bnstat[tid], red[tid]);
}

__global__ void bn_finalize(const float* __restrict__ bnstat, const float* __restrict__ g,
                            const float* __restrict__ bt, float* __restrict__ sc) {
  int i = threadIdx.x;
  if (i >= DIM) return;
  float inv = 1.f / (float)N_NODES;
  float mean = bnstat[i] * inv;
  float var = bnstat[DIM + i] * inv - mean * mean;
  float s = g[i] * rsqrtf(var + BN_EPS);
  sc[i] = s;
  sc[DIM + i] = bt[i] - mean * s;
}

__global__ void bn_apply(const float* __restrict__ r, const float* __restrict__ sc,
                         float* __restrict__ hout) {
  int idx = blockIdx.x * 256 + threadIdx.x;
  if (idx >= N_NODES * DIM) return;
  int i = idx & 31;
  hout[idx] = r[idx] * sc[i] + sc[DIM + i];
}

__global__ void pool_add(const float* __restrict__ h, const int* __restrict__ batch,
                         float* __restrict__ pooled) {
  int idx = blockIdx.x * 256 + threadIdx.x;
  if (idx >= N_NODES * DIM) return;
  atomicAdd(&pooled[(size_t)batch[idx >> 5] * DIM + (idx & 31)], h[idx]);
}

// ---------------------------------------------------------- protein branch
// Wt[c*256 + f*8 + k] = convW[f, c, k]  (coalesced layout for build_S)
__global__ void transpose_convw(const float* __restrict__ convW, float* __restrict__ Wt) {
  int idx = blockIdx.x * 256 + threadIdx.x;
  if (idx >= SEQ * 256) return;
  int c = idx >> 8, fk = idx & 255;
  Wt[idx] = convW[(size_t)(fk >> 3) * (SEQ * KW) + c * KW + (fk & 7)];
}

// S[g, v*256+f*8+k] = sum over positions c with target[g,c]==v of convW[f,c,k].
// One block per graph; 26x256 f32 accumulator in LDS; emit bf16 directly in
// A-fragment-swizzled order for the WMMA GEMM (M index = g, K index = i).
__global__ void build_S(const int* __restrict__ target, const float* __restrict__ Wt,
                        unsigned short* __restrict__ Ssw) {
  __shared__ float sS[VOCAB * 256];
  int g = blockIdx.x, tid = threadIdx.x;
  for (int i = tid; i < VOCAB * 256; i += 256) sS[i] = 0.f;
  __syncthreads();
  const int* tg = target + (size_t)g * SEQ;
  for (int c = 0; c < SEQ; ++c) {
    int v = tg[c];                           // uniform across block
    sS[v * 256 + tid] += Wt[c * 256 + tid];  // disjoint per-thread slots
  }
  __syncthreads();
  const int nkt = SROW >> 5;  // 208 k-tiles
  const int mt = g >> 4, rowin = g & 15;
  for (int i = tid; i < SROW; i += 256) {
    int kt = i >> 5, kk = i & 31;
    int half = (kk >> 3) & 1;                 // which half-wave holds this k
    int e = (kk & 7) + ((kk >> 4) << 3);      // element slot within lane
    int L = (half << 4) + rowin;
    size_t oidx = ((size_t)(mt * nkt + kt) * 32 + L) * 16 + e;
    Ssw[oidx] = f2bf(sS[i]);
  }
}

// Q[(v*256+f*8+k), o] = sum_t Emb[v, t+k] * Wxt[f*121+t, o]
__global__ void compute_Q(const float* __restrict__ Emb, const float* __restrict__ Wxt,
                          float* __restrict__ Q) {
  int idx = blockIdx.x * 256 + threadIdx.x;
  if (idx >= SROW * OD) return;
  int o = idx & (OD - 1);
  int r = idx >> 7;
  int v = r >> 8, f = (r >> 3) & 31, k = r & 7;
  const float* erow = Emb + v * 128 + k;
  const float* wcol = Wxt + (size_t)(f * CONV_LEN) * OD + o;
  float acc = 0.f;
  for (int t = 0; t < CONV_LEN; ++t) acc += erow[t] * wcol[(size_t)t * OD];
  Q[idx] = acc;
}

// bias_xt[o] = bxt[o] + sum_f convb[f] * sum_t Wxt[f*121+t, o]
__global__ void compute_bias_xt(const float* __restrict__ Wxt, const float* __restrict__ convb,
                                const float* __restrict__ bxt, float* __restrict__ bias) {
  int o = threadIdx.x;
  if (o >= OD) return;
  float acc = bxt[o];
  for (int f = 0; f < NF; ++f) {
    float s = 0.f;
    const float* base = Wxt + (size_t)(f * CONV_LEN) * OD + o;
    for (int t = 0; t < CONV_LEN; ++t) s += base[(size_t)t * OD];
    acc += convb[f] * s;
  }
  bias[o] = acc;
}

// out[g] = f2[g,:] . Wo + bo  (wave per row, shuffle reduce)
__global__ void head_out(const float* __restrict__ f2buf, const float* __restrict__ Wo,
                         const float* __restrict__ bo, float* __restrict__ out) {
  int g = (blockIdx.x << 3) + (threadIdx.x >> 5);
  int lane = threadIdx.x & 31;
  float s = 0.f;
  for (int j = lane; j < 256; j += 32) s += f2buf[(size_t)g * 256 + j] * Wo[j];
  for (int off = 16; off; off >>= 1) s += __shfl_xor(s, off, 32);
  if (lane == 0) out[g] = s + bo[0];
}

// ------------------------------------------------------------------ launch
extern "C" void kernel_launch(void* const* d_in, const int* in_sizes, int n_in, void* d_out,
                              int out_size, void* d_ws, size_t ws_size, hipStream_t stream) {
  (void)in_sizes; (void)n_in; (void)out_size; (void)ws_size;
  // Input order: setup_inputs() insertion order, params dict flattened in
  // insertion order: x, edge_attr, [47 params], edge_index, batch, target.
  const float* x = (const float*)d_in[0];
  const float* eattr = (const float*)d_in[1];
  const float* We = (const float*)d_in[2];
  const float* be = (const float*)d_in[3];
  const float* linW = (const float*)d_in[4];
  const float* linb = (const float*)d_in[5];
  const float *W1[5], *b1[5], *W2[5], *b2[5], *gam[5], *bet[5];
  for (int l = 0; l < 5; ++l) {
    int base = 6 + l * 6;
    W1[l] = (const float*)d_in[base + 0];
    b1[l] = (const float*)d_in[base + 1];
    W2[l] = (const float*)d_in[base + 2];
    b2[l] = (const float*)d_in[base + 3];
    gam[l] = (const float*)d_in[base + 4];
    bet[l] = (const float*)d_in[base + 5];
  }
  const float* Wxd = (const float*)d_in[36];
  const float* bxd = (const float*)d_in[37];
  const float* Emb = (const float*)d_in[38];
  const float* convW = (const float*)d_in[39];
  const float* convb = (const float*)d_in[40];
  const float* Wxt = (const float*)d_in[41];
  const float* bxt = (const float*)d_in[42];
  const float* Wf1 = (const float*)d_in[43];
  const float* bf1 = (const float*)d_in[44];
  const float* Wf2 = (const float*)d_in[45];
  const float* bf2 = (const float*)d_in[46];
  const float* Wo = (const float*)d_in[47];
  const float* bo = (const float*)d_in[48];
  const int* ei = (const int*)d_in[49];
  const int* batch = (const int*)d_in[50];
  const int* target = (const int*)d_in[51];

  char* ws = (char*)d_ws;
  size_t off = 0;
  auto alloc = [&](size_t bytes) -> char* {
    char* p = ws + off;
    off = (off + bytes + 255) & ~(size_t)255;
    return p;
  };
  float* agg = (float*)alloc((size_t)N_NODES * FXD * 4);
  float* hA = (float*)alloc((size_t)N_NODES * DIM * 4);
  float* hB = (float*)alloc((size_t)N_NODES * DIM * 4);
  float* WeL = (float*)alloc(FE * FXD * 4);
  float* beL = (float*)alloc(FXD * 4);
  float* bnstat = (float*)alloc(64 * 4);
  float* bnsc = (float*)alloc(64 * 4);
  float* pooled = (float*)alloc((size_t)G_GRAPHS * DIM * 4);
  unsigned short* poolSw = (unsigned short*)alloc((size_t)G_GRAPHS * DIM * 2);
  float* xc = (float*)alloc((size_t)G_GRAPHS * 256 * 4);
  unsigned short* xcSw = (unsigned short*)alloc((size_t)G_GRAPHS * 256 * 2);
  float* Wt = (float*)alloc((size_t)SEQ * 256 * 4);
  unsigned short* Ssw = (unsigned short*)alloc((size_t)G_GRAPHS * SROW * 2);
  float* Q = (float*)alloc((size_t)SROW * OD * 4);
  unsigned short* Qsw = (unsigned short*)alloc((size_t)SROW * OD * 2);
  float* biasxt = (float*)alloc(OD * 4);
  unsigned short* WxdSw = (unsigned short*)alloc((size_t)DIM * OD * 2);
  unsigned short* Wf1Sw = (unsigned short*)alloc((size_t)256 * 1024 * 2);
  unsigned short* Wf2Sw = (unsigned short*)alloc((size_t)1024 * 256 * 2);
  float* f1 = (float*)alloc((size_t)G_GRAPHS * 1024 * 4);
  unsigned short* f1Sw = (unsigned short*)alloc((size_t)G_GRAPHS * 1024 * 2);
  float* f2b = (float*)alloc((size_t)G_GRAPHS * 256 * 4);

  auto zero = [&](float* p, int n) { fill0<<<(n + 255) / 256, 256, 0, stream>>>(p, n); };

  // ---- GNN layer 0 (78-dim)
  prep_wel<<<1, 96, 0, stream>>>(We, be, linW, linb, WeL, beL);
  zero(agg, N_NODES * FXD);
  zero(bnstat, 64);
  edge_scatter<<<N_EDGES / 8, 256, 0, stream>>>(ei, eattr, x, WeL, beL, agg, FXD);
  node_mlp<FXD><<<(N_NODES + 255) / 256, 256, 0, stream>>>(x, agg, W1[0], b1[0], W2[0], b2[0], hA, bnstat);
  bn_finalize<<<1, 32, 0, stream>>>(bnstat, gam[0], bet[0], bnsc);
  bn_apply<<<(N_NODES * DIM + 255) / 256, 256, 0, stream>>>(hA, bnsc, hB);
  // ---- GNN layers 1..4 (32-dim)
  for (int l = 1; l < 5; ++l) {
    zero(agg, N_NODES * DIM);
    zero(bnstat, 64);
    edge_scatter<<<N_EDGES / 8, 256, 0, stream>>>(ei, eattr, hB, We, be, agg, DIM);
    node_mlp<DIM><<<(N_NODES + 255) / 256, 256, 0, stream>>>(hB, agg, W1[l], b1[l], W2[l], b2[l], hA, bnstat);
    bn_finalize<<<1, 32, 0, stream>>>(bnstat, gam[l], bet[l], bnsc);
    bn_apply<<<(N_NODES * DIM + 255) / 256, 256, 0, stream>>>(hA, bnsc, hB);
  }
  // ---- pooling + xg = relu(pooled @ Wxd + bxd) -> xc[:, 0:128]
  zero(pooled, G_GRAPHS * DIM);
  pool_add<<<(N_NODES * DIM + 255) / 256, 256, 0, stream>>>(hB, batch, pooled);
  swizzle_A<<<(G_GRAPHS * DIM + 255) / 256, 256, 0, stream>>>(pooled, poolSw, DIM, DIM, G_GRAPHS * DIM);
  swizzle_B<<<(DIM * OD + 255) / 256, 256, 0, stream>>>(Wxd, WxdSw, DIM, OD);
  gemm_bf16_wmma<true><<<dim3(OD / 128, G_GRAPHS / 64), 256, 0, stream>>>(
      poolSw, WxdSw, bxd, xc, DIM, OD, 256, 0);
  // ---- protein branch: xt = S @ Q + bias -> xc[:, 128:256]
  transpose_convw<<<(SEQ * 256 + 255) / 256, 256, 0, stream>>>(convW, Wt);
  build_S<<<G_GRAPHS, 256, 0, stream>>>(target, Wt, Ssw);
  compute_Q<<<(SROW * OD + 255) / 256, 256, 0, stream>>>(Emb, Wxt, Q);
  compute_bias_xt<<<1, OD, 0, stream>>>(Wxt, convb, bxt, biasxt);
  swizzle_B<<<(SROW * OD + 255) / 256, 256, 0, stream>>>(Q, Qsw, SROW, OD);
  gemm_bf16_wmma<false><<<dim3(OD / 128, G_GRAPHS / 64), 256, 0, stream>>>(
      Ssw, Qsw, biasxt, xc, SROW, OD, 256, 128);
  // ---- MLP head
  swizzle_A<<<(G_GRAPHS * 256 + 255) / 256, 256, 0, stream>>>(xc, xcSw, 256, 256, G_GRAPHS * 256);
  swizzle_B<<<(256 * 1024 + 255) / 256, 256, 0, stream>>>(Wf1, Wf1Sw, 256, 1024);
  gemm_bf16_wmma<true><<<dim3(1024 / 128, G_GRAPHS / 64), 256, 0, stream>>>(
      xcSw, Wf1Sw, bf1, f1, 256, 1024, 1024, 0);
  swizzle_A<<<(G_GRAPHS * 1024 + 255) / 256, 256, 0, stream>>>(f1, f1Sw, 1024, 1024, G_GRAPHS * 1024);
  swizzle_B<<<(1024 * 256 + 255) / 256, 256, 0, stream>>>(Wf2, Wf2Sw, 1024, 256);
  gemm_bf16_wmma<true><<<dim3(256 / 128, G_GRAPHS / 64), 256, 0, stream>>>(
      f1Sw, Wf2Sw, bf2, f2b, 1024, 256, 256, 0);
  head_out<<<G_GRAPHS / 8, 256, 0, stream>>>(f2b, Wo, bo, (float*)d_out);
}
